// FrameTransformerDecoder_57389353009723
// MI455X (gfx1250) — compile-verified, matrix operand order checked by reading, not run on hardware
//
#include <hip/hip_runtime.h>
#include <hip/hip_bf16.h>

// ---------------------------------------------------------------------------
// FrameTransformerDecoder on MI455X (gfx1250), wave32 + WMMA bf16->f32.
// ---------------------------------------------------------------------------

typedef __bf16 bf16;
typedef __attribute__((ext_vector_type(16))) __bf16 v16bf;
typedef __attribute__((ext_vector_type(8)))  __bf16 v8bf;
typedef __attribute__((ext_vector_type(8)))  float   v8f;

#define NB_    4
#define SEQ    1024      // CS == W
#define CDIM   1024      // BINS
#define FDIM   4096      // FF2
#define NHEAD  8
#define HDIM   128
#define MROWS  (NB_ * SEQ)   // 4096 rows for every (B*W, feat) GEMM

// ---------------------------------------------------------------------------
// WMMA helpers
// ---------------------------------------------------------------------------
__device__ __forceinline__ v8f wmma_bf16(v16bf a, v16bf b, v8f c) {
  // D = A(16x32 bf16) x B(32x16 bf16) + C(16x16 f32)
  return __builtin_amdgcn_wmma_f32_16x16x32_bf16(
      /*neg_a=*/false, a, /*neg_b=*/false, b, /*c_mod=*/(short)0, c,
      /*reuse_a=*/false, /*reuse_b=*/false);
}

// Load one 16x32 bf16 tile in the A/B register layout.
// lane L holds row (row0 + L%16); element e holds K = (e<8?0:16) + (L/16)*8 + e%8
// => two contiguous 16-byte segments per lane.
__device__ __forceinline__ v16bf ld_tile(const bf16* __restrict__ base, int row0,
                                         int ld, int col0) {
  const int lane = threadIdx.x & 31;
  const int h    = lane >> 4;
  const bf16* p  = base + (size_t)(row0 + (lane & 15)) * (size_t)ld
                        + (size_t)(col0 + h * 8);
  v8bf lo = *(const v8bf*)(p);
  v8bf hi = *(const v8bf*)(p + 16);
  v16bf r;
#pragma unroll
  for (int i = 0; i < 8; ++i) { r[i] = lo[i]; r[8 + i] = hi[i]; }
  return r;
}

// Same tile load, but from an f32 LDS array (softmax probabilities),
// converting to bf16 at load time.
__device__ __forceinline__ v16bf ld_tile_lds(const float* base, int ld, int col0) {
  const int lane = threadIdx.x & 31;
  const int h    = lane >> 4;
  const float* p = base + (size_t)(lane & 15) * (size_t)ld + col0 + h * 8;
  v16bf r;
#pragma unroll
  for (int i = 0; i < 8; ++i) { r[i] = (bf16)p[i]; r[8 + i] = (bf16)p[i + 16]; }
  return r;
}

// ---------------------------------------------------------------------------
// Generic GEMM: out = act(A(MxK,bf16) @ W(NxK,bf16)^T + bias) [+ res]
// block = 128 threads (4 waves); each wave owns a 32x64 C tile
// (2x4 grid of 16x16 WMMA tiles) => 8 WMMA per 12 b128-loads per k-step.
// grid = (N/256, M/32)
// ---------------------------------------------------------------------------
__global__ void __launch_bounds__(128)
gemm_kernel(const bf16* __restrict__ A, int lda,
            const bf16* __restrict__ Bw, int ldb,
            const float* __restrict__ bias,
            const float* __restrict__ res, long ldres,
            float* __restrict__ outf, long ldof,
            bf16*  __restrict__ outb, long ldob,
            int K, int act) {
  const int wave = threadIdx.x >> 5;
  const int lane = threadIdx.x & 31;
  const int row0 = blockIdx.y * 32;
  const int col0 = (blockIdx.x * 4 + wave) * 64;

  v8f acc[2][4] = {};
  for (int k0 = 0; k0 < K; k0 += 32) {
    if (k0 + 256 < K) {  // global_prefetch_b8: stream next K chunk
      __builtin_prefetch(A  + (size_t)(row0 + (lane & 15)) * lda + k0 + 256, 0, 1);
      __builtin_prefetch(Bw + (size_t)(col0 + (lane & 15)) * ldb + k0 + 256, 0, 1);
    }
    v16bf a0 = ld_tile(A, row0,      lda, k0);
    v16bf a1 = ld_tile(A, row0 + 16, lda, k0);
#pragma unroll
    for (int j = 0; j < 4; ++j) {
      v16bf b = ld_tile(Bw, col0 + 16 * j, ldb, k0);
      acc[0][j] = wmma_bf16(a0, b, acc[0][j]);
      acc[1][j] = wmma_bf16(a1, b, acc[1][j]);
    }
  }

#pragma unroll
  for (int i = 0; i < 2; ++i) {
#pragma unroll
    for (int j = 0; j < 4; ++j) {
      const int col   = col0 + 16 * j + (lane & 15);
      const int rbase = row0 + 16 * i + 8 * (lane >> 4);
      const float bv  = bias ? bias[col] : 0.0f;
#pragma unroll
      for (int e = 0; e < 8; ++e) {
        const int row = rbase + e;
        float r = acc[i][j][e] + bv;
        if (act == 1) r = (r > 0.0f) ? r : 0.01f * r;       // leaky_relu
        else if (act == 2) r = r / (1.0f + __expf(-r));     // silu
        if (res)  r += res[(size_t)row * (size_t)ldres + col];
        if (outf) outf[(size_t)row * (size_t)ldof + col] = r;
        if (outb) outb[(size_t)row * (size_t)ldob + col] = (bf16)r;
      }
    }
  }
}

// ---------------------------------------------------------------------------
// Fused multi-band attention for one (b, head, 16-query tile).
//  S[q,k] = ( Q[q,:]·K[k,:] + erT[q,:]·Q[k,:] ) / 32 ; softmax ; O = S·V
// grid = (SEQ/16, NHEAD, B), block = 128 (4 waves). 64KB LDS.
// q-side A tiles (Q[q0], erT[q0]) are hoisted into registers and reused
// across all key tiles handled by the wave.
// ---------------------------------------------------------------------------
__global__ void __launch_bounds__(128)
attn_kernel(const bf16* __restrict__ qb, const bf16* __restrict__ kb,
            const bf16* __restrict__ vt, const bf16* __restrict__ ert,
            bf16* __restrict__ ob) {
  __shared__ float S[16][SEQ];          // 64 KB

  const int qt   = blockIdx.x;
  const int n    = blockIdx.y;
  const int b    = blockIdx.z;
  const int wave = threadIdx.x >> 5;
  const int lane = threadIdx.x & 31;
  const int q0   = qt * 16;

  const bf16* Q  = qb + (size_t)b * SEQ * CDIM + n * HDIM;        // row w stride CDIM
  const bf16* Kp = kb + (size_t)b * SEQ * CDIM + n * HDIM;
  const bf16* Vt = vt + ((size_t)b * CDIM + n * HDIM) * SEQ;      // row d stride SEQ

  // ---- phase 1: scores (QK^T + pos^T), waves split the 64 key tiles ----
  v16bf aq[HDIM / 32], ae[HDIM / 32];
#pragma unroll
  for (int dc = 0; dc < HDIM / 32; ++dc) {
    aq[dc] = ld_tile(Q,   q0, CDIM, dc * 32);
    ae[dc] = ld_tile(ert, q0, HDIM, dc * 32);   // erT: (SEQ x HDIM)
  }
  for (int kt = wave; kt < SEQ / 16; kt += 4) {
    const int k0 = kt * 16;
    v8f acc = {};
#pragma unroll
    for (int dc = 0; dc < HDIM / 32; ++dc) {
      v16bf bk = ld_tile(Kp, k0, CDIM, dc * 32);
      acc = wmma_bf16(aq[dc], bk, acc);
      v16bf bq = ld_tile(Q,  k0, CDIM, dc * 32);
      acc = wmma_bf16(ae[dc], bq, acc);
    }
    const int col = k0 + (lane & 15);
    const int rb  = 8 * (lane >> 4);
#pragma unroll
    for (int e = 0; e < 8; ++e) S[rb + e][col] = acc[e] * (1.0f / 32.0f);
  }
  __syncthreads();

  // ---- phase 2: softmax, one wave per 4 rows ----
  for (int r = wave * 4; r < wave * 4 + 4; ++r) {
    float m = -1e30f;
    for (int c = lane; c < SEQ; c += 32) m = fmaxf(m, S[r][c]);
#pragma unroll
    for (int off = 16; off; off >>= 1) m = fmaxf(m, __shfl_xor(m, off, 32));
    float sum = 0.0f;
    for (int c = lane; c < SEQ; c += 32) {
      float ev = __expf(S[r][c] - m);
      S[r][c] = ev;
      sum += ev;
    }
#pragma unroll
    for (int off = 16; off; off >>= 1) sum += __shfl_xor(sum, off, 32);
    const float inv = 1.0f / sum;
    for (int c = lane; c < SEQ; c += 32) S[r][c] *= inv;
  }
  __syncthreads();

  // ---- phase 3: O = A(16xSEQ) @ V(SEQ x HDIM); waves split 8 d-chunks ----
#pragma unroll
  for (int t = 0; t < 2; ++t) {
    const int dt = wave * 2 + t;
    v8f acc = {};
    for (int kc = 0; kc < SEQ / 32; ++kc) {
      v16bf a  = ld_tile_lds(&S[0][0], SEQ, kc * 32);
      v16bf bv = ld_tile(Vt, dt * 16, SEQ, kc * 32);
      acc = wmma_bf16(a, bv, acc);
    }
    const int col = n * HDIM + dt * 16 + (lane & 15);
    const int rb  = 8 * (lane >> 4);
#pragma unroll
    for (int e = 0; e < 8; ++e)
      ob[((size_t)b * SEQ + q0 + rb + e) * CDIM + col] = (bf16)acc[e];
  }
}

// ---------------------------------------------------------------------------
// LayerNorm over last dim; one 256-thread block per row.
// ---------------------------------------------------------------------------
__global__ void __launch_bounds__(256)
ln_kernel(const float* __restrict__ x, const float* __restrict__ g,
          const float* __restrict__ bb, float* __restrict__ outf,
          bf16* __restrict__ outb, int cols) {
  const int row = blockIdx.x;
  const float* xr = x + (size_t)row * cols;
  float s = 0.0f, s2 = 0.0f;
  for (int c = threadIdx.x; c < cols; c += 256) {
    float v = xr[c];
    s += v; s2 += v * v;
  }
#pragma unroll
  for (int off = 16; off; off >>= 1) {
    s  += __shfl_down(s, off, 32);
    s2 += __shfl_down(s2, off, 32);
  }
  __shared__ float ss[8], ss2[8];
  const int wv = threadIdx.x >> 5, ln = threadIdx.x & 31;
  if (ln == 0) { ss[wv] = s; ss2[wv] = s2; }
  __syncthreads();
  if (threadIdx.x == 0) {
    float a = 0.0f, a2 = 0.0f;
#pragma unroll
    for (int i = 0; i < 8; ++i) { a += ss[i]; a2 += ss2[i]; }
    ss[0] = a; ss2[0] = a2;
  }
  __syncthreads();
  const float mean = ss[0] / (float)cols;
  const float var  = ss2[0] / (float)cols - mean * mean;
  const float rstd = rsqrtf(var + 1e-5f);
  for (int c = threadIdx.x; c < cols; c += 256) {
    float v = (xr[c] - mean) * rstd * g[c] + bb[c];
    if (outf) outf[(size_t)row * cols + c] = v;
    if (outb) outb[(size_t)row * cols + c] = (bf16)v;
  }
}

// ---------------------------------------------------------------------------
// Bottleneck: out[b,w,h] = sum_c x[b,c,h,w] * wgt[c]   (tiled transpose)
// grid = (SEQ/32, CDIM/32, B), block = 256 (32x8)
// ---------------------------------------------------------------------------
__global__ void __launch_bounds__(256)
bottleneck_kernel(const float* __restrict__ x, const float* __restrict__ wgt,
                  float* __restrict__ out, int nch) {
  __shared__ float tile[32][33];
  const int b  = blockIdx.z;
  const int h0 = blockIdx.y * 32;
  const int w0 = blockIdx.x * 32;
  const int tx = threadIdx.x & 31, ty = threadIdx.x >> 5;
#pragma unroll
  for (int i = 0; i < 4; ++i) {
    const int hh = ty + 8 * i;
    float acc = 0.0f;
    for (int c = 0; c < nch; ++c)
      acc += x[(((size_t)b * nch + c) * CDIM + (h0 + hh)) * SEQ + (w0 + tx)] * wgt[c];
    tile[hh][tx] = acc;
  }
  __syncthreads();
#pragma unroll
  for (int i = 0; i < 4; ++i) {
    const int ww = ty + 8 * i;
    out[((size_t)b * SEQ + (w0 + ww)) * CDIM + (h0 + tx)] = tile[tx][ww];
  }
}

// ---------------------------------------------------------------------------
// Batched transpose: per batch, src(rows x cols) -> dst(cols x rows)
// grid = (cols/32, rows/32, batch), block = 256.
// ---------------------------------------------------------------------------
template <typename TI, typename TO>
__global__ void __launch_bounds__(256)
transpose_kernel(const TI* __restrict__ src, TO* __restrict__ dst,
                 int rows, int cols) {
  __shared__ float tile[32][33];
  const size_t bo = (size_t)blockIdx.z * rows * cols;
  const int r0 = blockIdx.y * 32, c0 = blockIdx.x * 32;
  const int tx = threadIdx.x & 31, ty = threadIdx.x >> 5;
#pragma unroll
  for (int i = 0; i < 4; ++i)
    tile[ty + 8 * i][tx] =
        (float)src[bo + (size_t)(r0 + ty + 8 * i) * cols + (c0 + tx)];
  __syncthreads();
#pragma unroll
  for (int i = 0; i < 4; ++i)
    dst[bo + (size_t)(c0 + ty + 8 * i) * rows + (r0 + tx)] =
        (TO)tile[tx][ty + 8 * i];
}

// ---------------------------------------------------------------------------
// f32 -> bf16 elementwise convert
// ---------------------------------------------------------------------------
__global__ void __launch_bounds__(256)
cvt_kernel(const float* __restrict__ s, bf16* __restrict__ d, size_t n) {
  size_t i = (size_t)blockIdx.x * 256 + threadIdx.x;
  if (i < n) d[i] = (bf16)s[i];
}

// ---------------------------------------------------------------------------
// Causal depthwise conv along W: y[b,w,c] = sum_j x[b, w-(kw-1)+j, c]*wgt[c,j]
// ---------------------------------------------------------------------------
__global__ void __launch_bounds__(256)
dwconv_kernel(const bf16* __restrict__ x, const float* __restrict__ wgt,
              bf16* __restrict__ y, int C, int kw) {
  const size_t idx = (size_t)blockIdx.x * 256 + threadIdx.x;
  const size_t total = (size_t)NB_ * SEQ * C;
  if (idx >= total) return;
  const int c    = (int)(idx % C);
  const size_t bw = idx / C;
  const int wpos = (int)(bw % SEQ);
  const int b    = (int)(bw / SEQ);
  float acc = 0.0f;
  for (int j = 0; j < kw; ++j) {
    const int wi = wpos - (kw - 1) + j;
    if (wi >= 0)
      acc += (float)x[((size_t)b * SEQ + wi) * C + c] * wgt[c * kw + j];
  }
  y[idx] = (bf16)acc;
}

// ---------------------------------------------------------------------------
// Host-side orchestration
// ---------------------------------------------------------------------------
extern "C" void kernel_launch(void* const* d_in, const int* in_sizes, int n_in,
                              void* d_out, int out_size, void* d_ws, size_t ws_size,
                              hipStream_t stream) {
  (void)in_sizes; (void)n_in; (void)out_size; (void)ws_size;

  // ---- input indices (setup_inputs dict order, nested dicts flattened) ----
  const float* x_in   = (const float*)d_in[0];
  const float* mem_in = (const float*)d_in[1];
  const float* bw     = (const float*)d_in[2];
  const float* mbw    = (const float*)d_in[3];
  // attn sets: base 4 + 9*s : wq,bq,wk,bk,wv,bv,wo,bo,er
  const float* c1Ldw = (const float*)d_in[40];
  const float* c1Lpw = (const float*)d_in[41];
  const float* c1Rdw = (const float*)d_in[42];
  const float* c1Rpw = (const float*)d_in[43];
  const float* c2dw  = (const float*)d_in[44];
  const float* c2pw  = (const float*)d_in[45];
  const float* c3w   = (const float*)d_in[46];
  const float* c4w   = (const float*)d_in[47];
  auto norm_g = [&](int i) { return (const float*)d_in[48 + 2 * (i - 1)]; };
  auto norm_b = [&](int i) { return (const float*)d_in[49 + 2 * (i - 1)]; };

  // ---- workspace layout (bytes) ----
  char* ws = (char*)d_ws;
  const size_t MB = 1ull << 20;
  const size_t OFF_WATTN = 0;             // 4 sets x 4 mats x 2MB
  const size_t OFF_ERT   = 32 * MB;       // 4 x 256KB
  const size_t OFF_C1LPW = 33 * MB;       // 8MB
  const size_t OFF_C1RPW = 41 * MB;       // 2MB
  const size_t OFF_C2PW  = 43 * MB;       // 8MB
  const size_t OFF_C3    = 51 * MB;       // 8MB
  const size_t OFF_C4    = 59 * MB;       // 8MB
  const size_t OFF_XB    = 67 * MB;       // 16MB f32
  const size_t OFF_MB    = 83 * MB;       // 16MB f32
  const size_t OFF_MBBF  = 99 * MB;       // 8MB
  const size_t OFF_HBF   = 107 * MB;      // 8MB
  const size_t OFF_QBF   = 115 * MB;      // 8MB
  const size_t OFF_KBF   = 123 * MB;      // 8MB
  const size_t OFF_VBF   = 131 * MB;      // 8MB
  const size_t OFF_VT    = 139 * MB;      // 8MB
  const size_t OFF_OBF   = 147 * MB;      // 8MB
  const size_t OFF_DWS   = 155 * MB;      // 8MB
  const size_t OFF_HBIG  = 163 * MB;      // 64MB f32
  const size_t OFF_H3BF  = 227 * MB;      // 32MB
  const size_t OFF_DW2BF = 259 * MB;      // 32MB

  float* XB   = (float*)(ws + OFF_XB);
  float* MBf  = (float*)(ws + OFF_MB);
  float* HBIG = (float*)(ws + OFF_HBIG);
  bf16*  MBBF = (bf16*)(ws + OFF_MBBF);
  bf16*  HBF  = (bf16*)(ws + OFF_HBF);
  bf16*  QBF  = (bf16*)(ws + OFF_QBF);
  bf16*  KBF  = (bf16*)(ws + OFF_KBF);
  bf16*  VBF  = (bf16*)(ws + OFF_VBF);
  bf16*  VT   = (bf16*)(ws + OFF_VT);
  bf16*  OBF  = (bf16*)(ws + OFF_OBF);
  bf16*  DWS  = (bf16*)(ws + OFF_DWS);
  bf16*  H3BF = (bf16*)(ws + OFF_H3BF);
  bf16*  DW2  = (bf16*)(ws + OFF_DW2BF);

  auto cvt = [&](const float* s, bf16* d, size_t n) {
    cvt_kernel<<<dim3((unsigned)((n + 255) / 256)), 256, 0, stream>>>(s, d, n);
  };
  auto launch_gemm = [&](const bf16* A, int lda, const bf16* Bw, int ldb,
                         const float* bias, const float* res, long ldres,
                         float* outf, long ldof, bf16* outb, long ldob,
                         int K, int N, int act) {
    dim3 g(N / 256, MROWS / 32);
    gemm_kernel<<<g, 128, 0, stream>>>(A, lda, Bw, ldb, bias, res, ldres,
                                       outf, ldof, outb, ldob, K, act);
  };

  // ---- weight prep: f32 -> bf16 (and er transposed) ----
  bf16* WBF[4][4];  // [set][wq,wk,wv,wo]
  bf16* ERT[4];
  for (int s = 0; s < 4; ++s) {
    const int base = 4 + 9 * s;
    const int midx[4] = {0, 2, 4, 6};  // wq wk wv wo within the set
    for (int m = 0; m < 4; ++m) {
      WBF[s][m] = (bf16*)(ws + OFF_WATTN + (size_t)(s * 4 + m) * 2 * MB);
      cvt((const float*)d_in[base + midx[m]], WBF[s][m], (size_t)CDIM * CDIM);
    }
    ERT[s] = (bf16*)(ws + OFF_ERT + (size_t)s * (CDIM * HDIM * 2));
    // er: (HDIM x SEQ) f32 -> erT: (SEQ x HDIM) bf16
    transpose_kernel<float, bf16><<<dim3(SEQ / 32, HDIM / 32, 1), 256, 0, stream>>>(
        (const float*)d_in[base + 8], ERT[s], HDIM, SEQ);
  }
  bf16* C1LPW = (bf16*)(ws + OFF_C1LPW);
  bf16* C1RPW = (bf16*)(ws + OFF_C1RPW);
  bf16* C2PW  = (bf16*)(ws + OFF_C2PW);
  bf16* C3    = (bf16*)(ws + OFF_C3);
  bf16* C4    = (bf16*)(ws + OFF_C4);
  cvt(c1Lpw, C1LPW, (size_t)FDIM * CDIM);
  cvt(c1Rpw, C1RPW, (size_t)CDIM * CDIM);
  cvt(c2pw,  C2PW,  (size_t)CDIM * FDIM);
  cvt(c3w,   C3,    (size_t)FDIM * CDIM);
  cvt(c4w,   C4,    (size_t)CDIM * FDIM);

  // ---- bottlenecks ----
  bottleneck_kernel<<<dim3(SEQ / 32, CDIM / 32, NB_), 256, 0, stream>>>(x_in, bw, XB, 8);
  bottleneck_kernel<<<dim3(SEQ / 32, CDIM / 32, NB_), 256, 0, stream>>>(mem_in, mbw, MBf, 8);
  cvt(MBf, MBBF, (size_t)MROWS * CDIM);

  // ---- attention block helper ----
  auto attn_block = [&](const bf16* qsrc, const bf16* kvsrc, int s) {
    const int base = 4 + 9 * s;
    const float* bq = (const float*)d_in[base + 1];
    const float* bk = (const float*)d_in[base + 3];
    const float* bv = (const float*)d_in[base + 5];
    const float* bo = (const float*)d_in[base + 7];
    launch_gemm(qsrc,  CDIM, WBF[s][0], CDIM, bq, nullptr, 0, nullptr, 0, QBF, CDIM, CDIM, CDIM, 0);
    launch_gemm(kvsrc, CDIM, WBF[s][1], CDIM, bk, nullptr, 0, nullptr, 0, KBF, CDIM, CDIM, CDIM, 0);
    launch_gemm(kvsrc, CDIM, WBF[s][2], CDIM, bv, nullptr, 0, nullptr, 0, VBF, CDIM, CDIM, CDIM, 0);
    transpose_kernel<bf16, bf16><<<dim3(CDIM / 32, SEQ / 32, NB_), 256, 0, stream>>>(
        VBF, VT, SEQ, CDIM);
    attn_kernel<<<dim3(SEQ / 16, NHEAD, NB_), 128, 0, stream>>>(QBF, KBF, VT, ERT[s], OBF);
    launch_gemm(OBF, CDIM, WBF[s][3], CDIM, bo, XB, CDIM, XB, CDIM, nullptr, 0, CDIM, CDIM, 0);
  };

  // ---- block 1: sa1 + ea1 ----
  ln_kernel<<<MROWS, 256, 0, stream>>>(XB, norm_g(1), norm_b(1), nullptr, HBF, CDIM);
  attn_block(HBF, HBF,  0);   // sa1
  attn_block(HBF, MBBF, 1);   // ea1

  // ---- conv block ----
  ln_kernel<<<MROWS, 256, 0, stream>>>(XB, norm_g(2), norm_b(2), nullptr, HBF, CDIM);
  dwconv_kernel<<<(unsigned)(((size_t)MROWS * CDIM + 255) / 256), 256, 0, stream>>>(
      HBF, c1Ldw, DWS, CDIM, 6);
  launch_gemm(DWS, CDIM, C1LPW, CDIM, nullptr, nullptr, 0, HBIG, FDIM, nullptr, 0,
              CDIM, FDIM, 1);  // leaky_relu
  dwconv_kernel<<<(unsigned)(((size_t)MROWS * CDIM + 255) / 256), 256, 0, stream>>>(
      HBF, c1Rdw, DWS, CDIM, 4);
  launch_gemm(DWS, CDIM, C1RPW, CDIM, nullptr, HBIG, FDIM, HBIG, FDIM, nullptr, 0,
              CDIM, CDIM, 0);  // hR added into cols [0,1024) of HBIG
  ln_kernel<<<MROWS, 256, 0, stream>>>(HBIG, norm_g(3), norm_b(3), nullptr, H3BF, FDIM);
  dwconv_kernel<<<(unsigned)(((size_t)MROWS * FDIM + 255) / 256), 256, 0, stream>>>(
      H3BF, c2dw, DW2, FDIM, 4);
  launch_gemm(DW2, FDIM, C2PW, FDIM, nullptr, XB, CDIM, XB, CDIM, nullptr, 0,
              FDIM, CDIM, 0);

  // ---- block 2: sa2, ea2 ----
  ln_kernel<<<MROWS, 256, 0, stream>>>(XB, norm_g(4), norm_b(4), nullptr, HBF, CDIM);
  attn_block(HBF, HBF,  2);   // sa2
  ln_kernel<<<MROWS, 256, 0, stream>>>(XB, norm_g(5), norm_b(5), nullptr, HBF, CDIM);
  attn_block(HBF, MBBF, 3);   // ea2

  // ---- FFN: silu(h @ c3^T) @ c4^T ----
  ln_kernel<<<MROWS, 256, 0, stream>>>(XB, norm_g(6), norm_b(6), nullptr, HBF, CDIM);
  launch_gemm(HBF, CDIM, C3, CDIM, nullptr, nullptr, 0, nullptr, 0, H3BF, FDIM,
              CDIM, FDIM, 2);  // silu, bf16 out
  launch_gemm(H3BF, FDIM, C4, FDIM, nullptr, XB, CDIM, XB, CDIM, nullptr, 0,
              FDIM, CDIM, 0);

  // ---- final: out[b,0,h,w] = xb[b,w,h] ----
  transpose_kernel<float, float><<<dim3(CDIM / 32, SEQ / 32, NB_), 256, 0, stream>>>(
      XB, (float*)d_out, SEQ, CDIM);
}
